// GIN_24859270709378
// MI455X (gfx1250) — matrix operand order repaired and checked by reading, not compiled, not run on hardware
//
#include <hip/hip_runtime.h>
#include <hip/hip_bf16.h>
#include <math.h>

// ---------------------------------------------------------------------------
// GIN forward for MI455X (gfx1250, wave32, WMMA).
// Per layer: edge scatter-add -> WMMA GEMM1 (fused h+agg) -> column stats ->
// BN+LeakyReLU -> WMMA GEMM2 (+LeakyReLU) -> column stats -> BN+LeakyReLU ->
// segment pool.  Head: lin1 + BN + lin2 + log_softmax.
//
// GEMM: both operands staged in LDS as bf16 (converted once per block):
//   - A (activations, 16 rows x KC) row-major  -> 2 ds_load_b128 per fragment
//   - W (weights, 256 cols x KC) column-major  -> 2 ds_load_b128 per fragment
// Each wave owns 64 output columns: one A fragment feeds 4
// v_wmma_f32_16x16x32_bf16 per 32-wide K step.
// ---------------------------------------------------------------------------

typedef __attribute__((ext_vector_type(16))) __bf16 v16bf;
typedef __attribute__((ext_vector_type(8)))  float  v8f;

#define LEAK 0.01f
#define BN_EPS 1e-5f
#define KC 64                      // K chunk staged in LDS per barrier round

__device__ __forceinline__ __bf16 f2bf(float f) {
  union { float f; unsigned u; } v; v.f = f;
  unsigned r = v.u + 0x7FFFu + ((v.u >> 16) & 1u);   // round-to-nearest-even
  unsigned short h = (unsigned short)(r >> 16);
  __bf16 out;
  __builtin_memcpy(&out, &h, 2);
  return out;
}

__device__ __forceinline__ float lrelu(float v) { return v >= 0.0f ? v : LEAK * v; }

// ---------------------------------------------------------------------------
// Edge scatter-add:  agg[dst] += h[src]   (D floats per edge, 4 per thread)
// ---------------------------------------------------------------------------
__global__ void scatter_add_kernel(const float* __restrict__ h,
                                   const int* __restrict__ src,
                                   const int* __restrict__ dst,
                                   float* __restrict__ agg, int E, int D) {
  long t = (long)blockIdx.x * blockDim.x + threadIdx.x;
  int chunks = D >> 2;
  if (t >= (long)E * chunks) return;
  int e = (int)(t / chunks);
  int c = (int)(t % chunks) << 2;
  int s = src[e], d = dst[e];
  float4 v = *(const float4*)(h + (size_t)s * D + c);
  float* ap = agg + (size_t)d * D + c;
  atomicAdd(ap + 0, v.x);
  atomicAdd(ap + 1, v.y);
  atomicAdd(ap + 2, v.z);
  atomicAdd(ap + 3, v.w);
}

// ---------------------------------------------------------------------------
// WMMA GEMM:  out[M,NN] = act( (in (+ agg)) @ W + bias )
// block = 128 threads = 4 waves; block tile = 16 rows x 256 cols;
// wave w owns cols [64w, 64w+64) (4 accumulators).  grid = (M/16, NN/256).
// M%16==0, K%KC==0, NN%256==0 -> no divergence, EXEC all-ones for WMMA.
// ---------------------------------------------------------------------------
template <bool HAS_AGG>
__global__ void gemm_bf16_kernel(const float* __restrict__ in,
                                 const float* __restrict__ agg,
                                 const float* __restrict__ W,
                                 const float* __restrict__ bias,
                                 float* __restrict__ out,
                                 int M, int K, int NN, int act) {
  __shared__ __align__(16) __bf16 wldsT[256 * KC];  // [c][kk] col-major weights
  __shared__ __align__(16) __bf16 alds[16 * KC];    // [r][kk] activations

  const int tid  = threadIdx.x;
  const int lane = tid & 31;
  const int wave = tid >> 5;
  const int rowBase = blockIdx.x * 16;
  const int cb = blockIdx.y * 256;                  // block column base

  const int row16 = lane & 15;
  const int khalf = (lane >> 4) * 16;               // A K sub-offset (lanes 16-31)
  const int kbB   = (lane >> 4) * 16;               // B K sub-offset (lanes 16-31)

  v8f acc[4];
#pragma unroll
  for (int t = 0; t < 4; ++t) acc[t] = (v8f){0.f,0.f,0.f,0.f,0.f,0.f,0.f,0.f};

  for (int k0 = 0; k0 < K; k0 += KC) {
    // ---- stage W[k0:k0+KC, cb:cb+256] -> LDS transposed bf16 ----
    for (int idx = tid; idx < KC * 64; idx += 128) {
      int kk = idx >> 6;
      int cv = idx & 63;
      float4 w4 = *(const float4*)(W + (size_t)(k0 + kk) * NN + cb + cv * 4);
      int c = cv * 4;
      wldsT[(c + 0) * KC + kk] = f2bf(w4.x);
      wldsT[(c + 1) * KC + kk] = f2bf(w4.y);
      wldsT[(c + 2) * KC + kk] = f2bf(w4.z);
      wldsT[(c + 3) * KC + kk] = f2bf(w4.w);
    }
    // ---- stage A rows (fused h+agg) -> LDS row-major bf16 ----
    for (int idx = tid; idx < 16 * (KC / 4); idx += 128) {
      int r  = idx >> 4;            // KC/4 == 16
      int kk = (idx & 15) * 4;
      const float* p = in + (size_t)(rowBase + r) * K + k0 + kk;
      float4 a4 = *(const float4*)p;
      if (HAS_AGG) {
        const float* q = agg + (size_t)(rowBase + r) * K + k0 + kk;
        float4 g4 = *(const float4*)q;
        a4.x += g4.x; a4.y += g4.y; a4.z += g4.z; a4.w += g4.w;
      }
      alds[r * KC + kk + 0] = f2bf(a4.x);
      alds[r * KC + kk + 1] = f2bf(a4.y);
      alds[r * KC + kk + 2] = f2bf(a4.z);
      alds[r * KC + kk + 3] = f2bf(a4.w);
    }
    if (k0 + KC < K) __builtin_prefetch(in + (size_t)rowBase * K + k0 + KC, 0, 3);
    __syncthreads();

    // ---- compute: 2 K-steps of 32 per chunk ----
#pragma unroll
    for (int kk0 = 0; kk0 < KC; kk0 += 32) {
      // A fragment: one contiguous 16-element run (k = 16h + s, matches B)
      const __bf16* ap = alds + row16 * KC + kk0 + khalf;
      union { uint4 q[2]; v16bf v; } au;
      au.q[0] = *(const uint4*)(ap);        // k = kk0+khalf+0..7
      au.q[1] = *(const uint4*)(ap + 8);    // k = kk0+khalf+8..15
      v16bf a = au.v;

#pragma unroll
      for (int t = 0; t < 4; ++t) {
        int coll = wave * 64 + t * 16 + (lane & 15);
        v16bf b = *(const v16bf*)(wldsT + (size_t)coll * KC + kk0 + kbB);
        acc[t] = __builtin_amdgcn_wmma_f32_16x16x32_bf16(false, a, false, b,
                                                         (short)0, acc[t],
                                                         false, false);
      }
    }
    __syncthreads();
  }

  // ---- epilogue: C/D layout (reg r -> row r, lanes>=16 -> row+8) ----
  const int rowOff = rowBase + ((lane >> 4) * 8);
#pragma unroll
  for (int t = 0; t < 4; ++t) {
    int col = cb + wave * 64 + t * 16 + (lane & 15);
    float bi = bias[col];
#pragma unroll
    for (int r = 0; r < 8; ++r) {
      float v = acc[t][r] + bi;
      if (act) v = lrelu(v);
      out[(size_t)(rowOff + r) * NN + col] = v;
    }
  }
}

// ---------------------------------------------------------------------------
// Per-column sum / sum-of-squares (for batch-statistics BN).
// ---------------------------------------------------------------------------
__global__ void colstats_kernel(const float* __restrict__ x,
                                float* __restrict__ stats,  // [2*C]
                                int M, int C, int rowsPerBlock) {
  int c = threadIdx.x;            // C threads per block
  int r0 = blockIdx.x * rowsPerBlock;
  int r1 = r0 + rowsPerBlock; if (r1 > M) r1 = M;
  float s = 0.f, s2 = 0.f;
  for (int r = r0; r < r1; ++r) {
    float v = x[(size_t)r * C + c];
    s += v; s2 += v * v;
  }
  atomicAdd(&stats[c], s);
  atomicAdd(&stats[C + c], s2);
}

// ---------------------------------------------------------------------------
// y = lrelu( (x - mean) * rsqrt(var+eps) * g + b ), biased var (jnp default)
// ---------------------------------------------------------------------------
__global__ void bn_lrelu_kernel(const float* __restrict__ x,
                                float* __restrict__ y,
                                const float* __restrict__ stats,
                                const float* __restrict__ g,
                                const float* __restrict__ b,
                                int M, int C) {
  long idx = (long)blockIdx.x * blockDim.x + threadIdx.x;
  if (idx >= (long)M * C) return;
  int c = (int)(idx % C);
  float invM = 1.0f / (float)M;
  float mean = stats[c] * invM;
  float var  = stats[C + c] * invM - mean * mean;
  float v = (x[idx] - mean) * rsqrtf(var + BN_EPS) * g[c] + b[c];
  y[idx] = lrelu(v);
}

// ---------------------------------------------------------------------------
// Segment pooling:  pools[batch[r], colOff + c] += h[r, c]
// ---------------------------------------------------------------------------
__global__ void pool_add_kernel(const float* __restrict__ h,
                                const int* __restrict__ batch,
                                float* __restrict__ pools,
                                int M, int C, int poolStride, int colOff) {
  long idx = (long)blockIdx.x * blockDim.x + threadIdx.x;
  if (idx >= (long)M * C) return;
  int r = (int)(idx / C), c = (int)(idx % C);
  atomicAdd(&pools[(size_t)batch[r] * poolStride + colOff + c], h[idx]);
}

// ---------------------------------------------------------------------------
// h1[g,j] = lrelu( concat(pools[g], kmers[g]) @ lin1_W[:,j] + lin1_b[j] )
// ---------------------------------------------------------------------------
__global__ void lin1_kernel(const float* __restrict__ pools,
                            const float* __restrict__ kmers,
                            const float* __restrict__ W,   // [832,64]
                            const float* __restrict__ bias,
                            float* __restrict__ h1) {
  int g = blockIdx.x;      // 512
  int j = threadIdx.x;     // 64
  float acc = bias[j];
  const float* p = pools + (size_t)g * 768;
  for (int c = 0; c < 768; ++c) acc += p[c] * W[c * 64 + j];
  const float* km = kmers + (size_t)g * 64;
  for (int c = 0; c < 64; ++c) acc += km[c] * W[(768 + c) * 64 + j];
  h1[(size_t)g * 64 + j] = lrelu(acc);
}

// mean/var per column of h1 [512,64] -> st[0..63]=mean, st[64..127]=var
__global__ void stats64_kernel(const float* __restrict__ h1,
                               float* __restrict__ st) {
  int j = threadIdx.x;  // 64
  float s = 0.f, s2 = 0.f;
  for (int g = 0; g < 512; ++g) {
    float v = h1[(size_t)g * 64 + j];
    s += v; s2 += v * v;
  }
  float mean = s * (1.0f / 512.0f);
  st[j] = mean;
  st[64 + j] = s2 * (1.0f / 512.0f) - mean * mean;
}

// BN(h1) @ lin2_W + lin2_b, then log_softmax; d_out = [out | logsm] flat.
__global__ void head_kernel(const float* __restrict__ h1,
                            const float* __restrict__ st,
                            const float* __restrict__ bg,
                            const float* __restrict__ bb,
                            const float* __restrict__ W2,  // [64,2]
                            const float* __restrict__ b2,
                            float* __restrict__ out) {
  int g = blockIdx.x * blockDim.x + threadIdx.x;
  if (g >= 512) return;
  float o0 = b2[0], o1 = b2[1];
  for (int j = 0; j < 64; ++j) {
    float v = (h1[(size_t)g * 64 + j] - st[j]) * rsqrtf(st[64 + j] + BN_EPS) * bg[j] + bb[j];
    o0 += v * W2[j * 2 + 0];
    o1 += v * W2[j * 2 + 1];
  }
  out[g * 2 + 0] = o0;
  out[g * 2 + 1] = o1;
  float m = fmaxf(o0, o1);
  float lse = m + logf(expf(o0 - m) + expf(o1 - m));
  out[1024 + g * 2 + 0] = o0 - lse;
  out[1024 + g * 2 + 1] = o1 - lse;
}

// ---------------------------------------------------------------------------
extern "C" void kernel_launch(void* const* d_in, const int* in_sizes, int n_in,
                              void* d_out, int out_size, void* d_ws, size_t ws_size,
                              hipStream_t stream) {
  const int N = 50000, E = 800000, G = 512, HID = 64, DFF = 256;

  const float* x      = (const float*)d_in[0];
  const int*   ei     = (const int*)  d_in[1];
  const int*   src    = ei;
  const int*   dst    = ei + E;
  const int*   batch  = (const int*)  d_in[2];
  const float* kmers  = (const float*)d_in[3];
  const float* l0_W1  = (const float*)d_in[4];
  const float* l0_b1  = (const float*)d_in[5];
  const float* l0_g1  = (const float*)d_in[6];
  const float* l0_bt1 = (const float*)d_in[7];
  const float* l0_W2  = (const float*)d_in[8];
  const float* l0_b2  = (const float*)d_in[9];
  const float* l0_go  = (const float*)d_in[10];
  const float* l0_bto = (const float*)d_in[11];
  const float* Ws1    = (const float*)d_in[12];
  const float* bs1    = (const float*)d_in[13];
  const float* gs1    = (const float*)d_in[14];
  const float* bts1   = (const float*)d_in[15];
  const float* Ws2    = (const float*)d_in[16];
  const float* bs2    = (const float*)d_in[17];
  const float* gso    = (const float*)d_in[18];
  const float* btso   = (const float*)d_in[19];
  const float* lin1_W = (const float*)d_in[20];
  const float* lin1_b = (const float*)d_in[21];
  const float* bn1_g  = (const float*)d_in[22];
  const float* bn1_b  = (const float*)d_in[23];
  const float* lin2_W = (const float*)d_in[24];
  const float* lin2_b = (const float*)d_in[25];

  // workspace carve (floats)
  float* h     = (float*)d_ws;                    // N*DFF
  float* agg   = h     + (size_t)N * DFF;         // N*DFF (also reused as t2)
  float* tmp   = agg   + (size_t)N * DFF;         // N*DFF
  float* pools = tmp   + (size_t)N * DFF;         // G*768
  float* stats = pools + (size_t)G * 768;         // 2*DFF
  float* h1    = stats + 2 * DFF;                 // G*64
  float* st2   = h1    + (size_t)G * 64;          // 128

  const dim3 gemmGrid(N / 16, DFF / 256);
  const int  gemmBlock = 128;
  const long elemsND = (long)N * DFF;
  const int  elemGrid = (int)((elemsND + 255) / 256);
  const int  statGrid = (N + 255) / 256;

  hipMemsetAsync(pools, 0, (size_t)G * 768 * sizeof(float), stream);

  // ---------------- layer 0 (input dim 64) ----------------
  hipMemsetAsync(agg, 0, (size_t)N * HID * sizeof(float), stream);
  {
    long total = (long)E * (HID / 4);
    scatter_add_kernel<<<(int)((total + 255) / 256), 256, 0, stream>>>(x, src, dst, agg, E, HID);
  }
  gemm_bf16_kernel<true><<<gemmGrid, gemmBlock, 0, stream>>>(x, agg, l0_W1, l0_b1, tmp, N, HID, DFF, 0);
  hipMemsetAsync(stats, 0, 2 * DFF * sizeof(float), stream);
  colstats_kernel<<<statGrid, DFF, 0, stream>>>(tmp, stats, N, DFF, 256);
  bn_lrelu_kernel<<<elemGrid, 256, 0, stream>>>(tmp, tmp, stats, l0_g1, l0_bt1, N, DFF);
  gemm_bf16_kernel<false><<<gemmGrid, gemmBlock, 0, stream>>>(tmp, nullptr, l0_W2, l0_b2, agg, N, DFF, DFF, 1);
  hipMemsetAsync(stats, 0, 2 * DFF * sizeof(float), stream);
  colstats_kernel<<<statGrid, DFF, 0, stream>>>(agg, stats, N, DFF, 256);
  bn_lrelu_kernel<<<elemGrid, 256, 0, stream>>>(agg, h, stats, l0_go, l0_bto, N, DFF);
  pool_add_kernel<<<elemGrid, 256, 0, stream>>>(h, batch, pools, N, DFF, 768, 0);

  // ---------------- layers 1, 2 (dim 256) ----------------
  for (int i = 0; i < 2; ++i) {
    const float* W1 = Ws1  + (size_t)i * DFF * DFF;
    const float* b1 = bs1  + (size_t)i * DFF;
    const float* g1 = gs1  + (size_t)i * DFF;
    const float* t1 = bts1 + (size_t)i * DFF;
    const float* W2 = Ws2  + (size_t)i * DFF * DFF;
    const float* b2 = bs2  + (size_t)i * DFF;
    const float* go = gso  + (size_t)i * DFF;
    const float* to = btso + (size_t)i * DFF;

    hipMemsetAsync(agg, 0, (size_t)N * DFF * sizeof(float), stream);
    {
      long total = (long)E * (DFF / 4);
      scatter_add_kernel<<<(int)((total + 255) / 256), 256, 0, stream>>>(h, src, dst, agg, E, DFF);
    }
    gemm_bf16_kernel<true><<<gemmGrid, gemmBlock, 0, stream>>>(h, agg, W1, b1, tmp, N, DFF, DFF, 0);
    hipMemsetAsync(stats, 0, 2 * DFF * sizeof(float), stream);
    colstats_kernel<<<statGrid, DFF, 0, stream>>>(tmp, stats, N, DFF, 256);
    bn_lrelu_kernel<<<elemGrid, 256, 0, stream>>>(tmp, tmp, stats, g1, t1, N, DFF);
    gemm_bf16_kernel<false><<<gemmGrid, gemmBlock, 0, stream>>>(tmp, nullptr, W2, b2, agg, N, DFF, DFF, 1);
    hipMemsetAsync(stats, 0, 2 * DFF * sizeof(float), stream);
    colstats_kernel<<<statGrid, DFF, 0, stream>>>(agg, stats, N, DFF, 256);
    bn_lrelu_kernel<<<elemGrid, 256, 0, stream>>>(agg, h, stats, go, to, N, DFF);
    pool_add_kernel<<<elemGrid, 256, 0, stream>>>(h, batch, pools, N, DFF, 768, (i + 1) * DFF);
  }

  // ---------------- head ----------------
  lin1_kernel<<<G, 64, 0, stream>>>(pools, kmers, lin1_W, lin1_b, h1);
  stats64_kernel<<<1, 64, 0, stream>>>(h1, st2);
  head_kernel<<<(G + 63) / 64, 64, 0, stream>>>(h1, st2, bn1_g, bn1_b, lin2_W, lin2_b,
                                                (float*)d_out);
}
